// PacketMambaLayer_25357486916253
// MI455X (gfx1250) — matrix operand order repaired and checked by reading, not compiled, not run on hardware
//
#include <hip/hip_runtime.h>
#include <hip/hip_bf16.h>
#include <math.h>

#define DEV static __device__ __forceinline__

typedef __bf16 bf16;
typedef __attribute__((ext_vector_type(16))) __bf16 v16bf;
typedef __attribute__((ext_vector_type(8)))  __bf16 v8bf;
typedef __attribute__((ext_vector_type(8)))  float  v8f;

static constexpr int Bq = 2, Nn = 512, Dm = 128, DSt = 16;
static constexpr int ROWS = Bq * Nn;            // 1024
static constexpr int BND  = Bq * Nn * Dm;       // 131072
static constexpr int EDGE_SMEM = 175168;        // bytes of dynamic LDS for k_edge

DEV float geluf(float x)  { return 0.5f * x * (1.0f + erff(x * 0.7071067811865476f)); }
DEV float sigmf(float x)  { return 1.0f / (1.0f + expf(-x)); }
DEV float softpf(float x) { return (x > 20.0f) ? x : log1pf(expf(x)); }
DEV bf16  tob(float x)    { return (bf16)x; }

DEV v16bf frag16(const bf16* p0, const bf16* p1) {
  v8bf lo = *(const v8bf*)p0;
  v8bf hi = *(const v8bf*)p1;
  return __builtin_shufflevector(lo, hi, 0,1,2,3,4,5,6,7,8,9,10,11,12,13,14,15);
}
// A 16x32 bf16 fragment: lane<16 row=lane holds K{0..7,16..23}; lane>=16 holds K{8..15,24..31}
DEV v16bf a_frag(const bf16* row, int hi) { return frag16(row + hi*8, row + 16 + hi*8); }
// B 32x16 bf16 fragment from column-major (N-major) tile: lane<16 col=lane holds K{0..7,16..23}; lane>=16 holds K{8..15,24..31}
DEV v16bf b_frag(const bf16* colrow, int hi) { return frag16(colrow + hi*8, colrow + 16 + hi*8); }

// ---------------- LayerNorm: 1 wave per 128-wide row ----------------
__global__ __launch_bounds__(256) void k_ln(const float* __restrict__ in, const float* __restrict__ g,
                                            const float* __restrict__ b, float* __restrict__ out, int rows) {
  int wv = threadIdx.x >> 5, lane = threadIdx.x & 31;
  int r = blockIdx.x * 8 + wv;
  if (r >= rows) return;
  const float4 x4 = *(const float4*)(in + (size_t)r * Dm + lane * 4);
  float s  = x4.x + x4.y + x4.z + x4.w;
  float ss = x4.x*x4.x + x4.y*x4.y + x4.z*x4.z + x4.w*x4.w;
  for (int o = 16; o; o >>= 1) { s += __shfl_xor(s, o); ss += __shfl_xor(ss, o); }
  float m = s * (1.0f / Dm);
  float v = ss * (1.0f / Dm) - m * m;
  float rstd = rsqrtf(v + 1e-5f);
  int c = lane * 4;
  float4 o4;
  o4.x = (x4.x - m) * rstd * g[c+0] + b[c+0];
  o4.y = (x4.y - m) * rstd * g[c+1] + b[c+1];
  o4.z = (x4.z - m) * rstd * g[c+2] + b[c+2];
  o4.w = (x4.w - m) * rstd * g[c+3] + b[c+3];
  *(float4*)(out + (size_t)r * Dm + c) = o4;
}

// ------------- Generic bf16-WMMA GEMM: C = act(A@W + bias + add) -------------
// 64x64 block tile, 8 waves (4 M-subtiles x 2 N-subtiles of 16x32). act: 0=none 1=gelu 2=sigmoid 3=softplus+1e-3
__global__ __launch_bounds__(256) void k_gemm(const float* __restrict__ A, int lda,
                                              const float* __restrict__ W, int ldw,
                                              const float* __restrict__ bias,
                                              const float* __restrict__ add,
                                              float* __restrict__ C,
                                              int M, int N, int K, int act) {
  __shared__ bf16 sA [64 * 40];   // [row][k], pitch 40 halves (80B, 16B-aligned rows)
  __shared__ bf16 sWt[64 * 40];   // [n][k] transposed
  const int t = threadIdx.x, lane = t & 31, wv = t >> 5;
  const int mw = wv & 3, nw = wv >> 2;
  const int m0 = blockIdx.y * 64, n0 = blockIdx.x * 64;
  const int row = lane & 15, hi = lane >> 4;
  v8f acc0 = {0,0,0,0,0,0,0,0}, acc1 = {0,0,0,0,0,0,0,0};
  for (int kb = 0; kb < K; kb += 32) {
    for (int idx = t; idx < 64 * 32; idx += 256) {
      int r = idx >> 5, kk = idx & 31;
      sA[r * 40 + kk] = tob(A[(size_t)(m0 + r) * lda + kb + kk]);
    }
    for (int idx = t; idx < 32 * 64; idx += 256) {
      int k = idx >> 6, n = idx & 63;
      sWt[n * 40 + k] = tob(W[(size_t)(kb + k) * ldw + n0 + n]);
    }
    __syncthreads();
    v16bf af  = a_frag(sA + (mw * 16 + row) * 40, hi);
    v16bf bf0 = b_frag(sWt + ((nw * 2 + 0) * 16 + row) * 40, hi);
    acc0 = __builtin_amdgcn_wmma_f32_16x16x32_bf16(false, af, false, bf0, (short)0, acc0, false, false);
    v16bf bf1 = b_frag(sWt + ((nw * 2 + 1) * 16 + row) * 40, hi);
    acc1 = __builtin_amdgcn_wmma_f32_16x16x32_bf16(false, af, false, bf1, (short)0, acc1, false, false);
    __syncthreads();
  }
  for (int j = 0; j < 2; ++j) {
    v8f a = j ? acc1 : acc0;
    int n = n0 + (nw * 2 + j) * 16 + row;
    float bv = bias ? bias[n] : 0.0f;
    for (int r = 0; r < 8; ++r) {
      int m = m0 + mw * 16 + r + hi * 8;
      float v = a[r] + bv;
      if (add) v += add[(size_t)m * N + n];
      if      (act == 1) v = geluf(v);
      else if (act == 2) v = sigmf(v);
      else if (act == 3) v = softpf(v) + 1e-3f;
      C[(size_t)m * N + n] = v;
    }
  }
}

// ------------- Fused dense edge mixer message kernel -------------
// grid = B * (N/16), block = 256. Produces messages[b,i,:] = sum_j mask*sigmoid(g2(gelu(h)))*src[j] / deg[i]
__global__ __launch_bounds__(256) void k_edge(const float* __restrict__ prei, const float* __restrict__ prej,
                                              const float* __restrict__ srcb, const float* __restrict__ ef,
                                              const unsigned char* __restrict__ adj,
                                              const unsigned char* __restrict__ nmask,
                                              const float* __restrict__ w1e,   // em_g1_w rows 256..259  [4][128]
                                              const float* __restrict__ w2,    // em_g2_w [128][128]
                                              const float* __restrict__ b2,    // em_g2_b [128]
                                              float* __restrict__ messages) {
  extern __shared__ char smem[];
  bf16*  sW2t = (bf16*)(smem + 0);        // [128][136] W2 transposed (n-major)
  float* sW1e = (float*)(smem + 34816);   // [4][128]
  float* sPi  = (float*)(smem + 36864);   // [16][128]
  float* sMsg = (float*)(smem + 45056);   // [16][128]
  float* sDeg = (float*)(smem + 53248);   // [16]
  float* sPj  = (float*)(smem + 53312);   // [16][128]
  float* sSrc = (float*)(smem + 61504);   // [16][128]
  float* sEf  = (float*)(smem + 69696);   // [16][16][4]
  float* sMk  = (float*)(smem + 73792);   // [16][16]
  bf16*  sH   = (bf16*)(smem + 74816);    // [128][136]  (128 edge-pairs x 128 ch, bf16)
  float* sG   = (float*)(smem + 109632);  // [128][128]  gates f32

  const int b = blockIdx.x >> 5, i0 = (blockIdx.x & 31) * 16;
  const int t = threadIdx.x, lane = t & 31, wv = t >> 5;
  const int row = lane & 15, hi = lane >> 4;

  for (int idx = t; idx < Dm * Dm; idx += 256) { int k = idx >> 7, n = idx & 127; sW2t[n * 136 + k] = tob(w2[idx]); }
  for (int idx = t; idx < 4 * Dm;  idx += 256) sW1e[idx] = w1e[idx];
  for (int idx = t; idx < 16 * Dm; idx += 256)
    sPi[idx] = prei[(size_t)(b * Nn + i0 + (idx >> 7)) * Dm + (idx & 127)];
  for (int idx = t; idx < 16 * Dm; idx += 256) sMsg[idx] = 0.0f;
  if (t < 16) {
    float dg = 0.0f;
    const unsigned char* ar = adj + (size_t)(b * Nn + i0 + t) * Nn;
    for (int j = 0; j < Nn; ++j) dg += ar[j] ? 1.0f : 0.0f;
    sDeg[t] = fmaxf(dg, 1.0f);
  }
  __syncthreads();

  for (int jt = 0; jt < 32; ++jt) {
    const int j0 = jt * 16;
    for (int idx = t; idx < 16 * Dm; idx += 256) {
      int r = idx >> 7, c = idx & 127;
      size_t g = (size_t)(b * Nn + j0 + r) * Dm + c;
      sPj[idx]  = prej[g];
      sSrc[idx] = srcb[g];
    }
    for (int idx = t; idx < 16 * 16 * 4; idx += 256) {
      int ii = idx >> 6, jj = (idx >> 2) & 15, e = idx & 3;
      sEf[idx] = ef[((size_t)(b * Nn + i0 + ii) * Nn + j0 + jj) * 4 + e];
    }
    if (t < 256) {
      int ii = t >> 4, jj = t & 15;
      bool am = adj[(size_t)(b * Nn + i0 + ii) * Nn + j0 + jj] != 0;
      bool nm = nmask[b * Nn + j0 + jj] != 0;
      sMk[t] = (am && nm) ? 1.0f : 0.0f;
    }
    __syncthreads();

    for (int half = 0; half < 2; ++half) {
      // build h = gelu(prei[i] + prej[j] + ef@W1e) for 128 (ii,jj) pairs, bf16 into LDS
      for (int idx = t; idx < 128 * Dm; idx += 256) {
        int p = idx >> 7, d = idx & 127;
        int iiL = p >> 4, jj = p & 15, ii = half * 8 + iiL;
        const float* e4 = sEf + (ii * 16 + jj) * 4;
        float v = sPi[ii * Dm + d] + sPj[jj * Dm + d]
                + e4[0] * sW1e[d]       + e4[1] * sW1e[128 + d]
                + e4[2] * sW1e[256 + d] + e4[3] * sW1e[384 + d];
        sH[p * 136 + d] = tob(geluf(v));
      }
      __syncthreads();
      // WMMA GEMM: [128 pairs x 128] @ [128 x 128]; wave = one 16-row M-tile
      v8f acc[8];
      for (int nt = 0; nt < 8; ++nt) for (int r = 0; r < 8; ++r) acc[nt][r] = 0.0f;
      const bf16* aRow = sH + (wv * 16 + row) * 136;
      for (int ks = 0; ks < 4; ++ks) {
        v16bf af = a_frag(aRow + ks * 32, hi);
        for (int nt = 0; nt < 8; ++nt) {
          v16bf bfv = b_frag(sW2t + (nt * 16 + row) * 136 + ks * 32, hi);
          acc[nt] = __builtin_amdgcn_wmma_f32_16x16x32_bf16(false, af, false, bfv, (short)0, acc[nt], false, false);
        }
      }
      // sigmoid(acc + bias) -> gates (f32 LDS)
      for (int nt = 0; nt < 8; ++nt) {
        int n = nt * 16 + row;
        float bv = b2[n];
        for (int r = 0; r < 8; ++r) {
          int p = wv * 16 + r + hi * 8;
          sG[p * Dm + n] = sigmf(acc[nt][r] + bv);
        }
      }
      __syncthreads();
      // deterministic owner-computes masked accumulation into sMsg
      {
        int iiL = t >> 5, ii = half * 8 + iiL;
        int c0 = (t & 31) * 4;
        float a0 = 0, a1 = 0, a2 = 0, a3 = 0;
        for (int jj = 0; jj < 16; ++jj) {
          float mk = sMk[ii * 16 + jj];
          const float* gp = sG + (iiL * 16 + jj) * Dm + c0;
          const float* sp = sSrc + jj * Dm + c0;
          a0 += mk * gp[0] * sp[0];
          a1 += mk * gp[1] * sp[1];
          a2 += mk * gp[2] * sp[2];
          a3 += mk * gp[3] * sp[3];
        }
        sMsg[ii * Dm + c0 + 0] += a0;
        sMsg[ii * Dm + c0 + 1] += a1;
        sMsg[ii * Dm + c0 + 2] += a2;
        sMsg[ii * Dm + c0 + 3] += a3;
      }
      __syncthreads();
    }
  }
  for (int idx = t; idx < 16 * Dm; idx += 256) {
    int r = idx >> 7, c = idx & 127;
    messages[(size_t)(b * Nn + i0 + r) * Dm + c] = sMsg[idx] / sDeg[r];
  }
}

// ------------- Mamba-like selective scan: 1 thread per (b,d) channel -------------
__global__ void k_scan(const float* __restrict__ delta, const float* __restrict__ bt,
                       const float* __restrict__ ct, const float* __restrict__ gate,
                       const float* __restrict__ nrm, const float* __restrict__ skip,
                       const float* __restrict__ log_a, const unsigned char* __restrict__ smask,
                       float* __restrict__ yg) {
  int id = blockIdx.x * 32 + threadIdx.x;   // 0..255
  int b = id >> 7, d = id & 127;
  float av[DSt], binv[DSt], st[DSt];
  for (int s = 0; s < DSt; ++s) {
    float a = -softpf(log_a[d * DSt + s]) - 1e-4f;
    av[s] = a; binv[s] = 1.0f / (-a + 1e-4f); st[s] = 0.0f;
  }
  float sk = skip[d];
  for (int tt = 0; tt < Nn; ++tt) {
    size_t ridx = (size_t)(b * Nn + tt);
    float dt = delta[ridx * Dm + d];
    bool m = smask[ridx] != 0;
    const float* bp = bt + (ridx * Dm + d) * DSt;
    const float* cp = ct + (ridx * Dm + d) * DSt;
    float y = 0.0f;
    for (int s = 0; s < DSt; ++s) {
      float al = expf(dt * av[s]);
      float be = (1.0f - al) * binv[s];
      float up = al * st[s] + be * bp[s];
      if (m) st[s] = up;
      y += cp[s] * st[s];
    }
    yg[ridx * Dm + d] = gate[ridx * Dm + d] * (y + sk * nrm[ridx * Dm + d]);
  }
}

__global__ void k_gather(const float* __restrict__ in, const int* __restrict__ order, float* __restrict__ out) {
  int i = blockIdx.x * 256 + threadIdx.x;
  int c = i & 127, n = (i >> 7) & 511, b = i >> 16;
  int src = order[b * Nn + n];
  out[i] = in[((size_t)b * Nn + src) * Dm + c];
}

__global__ void k_scatter(const float* __restrict__ in, const int* __restrict__ order, float* __restrict__ out) {
  int i = blockIdx.x * 256 + threadIdx.x;
  int c = i & 127, n = (i >> 7) & 511, b = i >> 16;
  int dst = order[b * Nn + n];
  out[((size_t)b * Nn + dst) * Dm + c] = in[i];
}

__global__ void k_combine(const float* __restrict__ x1, const float* __restrict__ mg,
                          const float* __restrict__ cur, const float* __restrict__ dst,
                          float* __restrict__ out) {
  int i = blockIdx.x * 256 + threadIdx.x;
  float m = mg[i];
  out[i] = x1[i] + m * cur[i] + (1.0f - m) * dst[i];
}

extern "C" void kernel_launch(void* const* d_in, const int* in_sizes, int n_in,
                              void* d_out, int out_size, void* d_ws, size_t ws_size,
                              hipStream_t stream) {
  (void)in_sizes; (void)n_in; (void)out_size; (void)ws_size;
  const float* x      = (const float*)d_in[0];
  const float* ef     = (const float*)d_in[1];
  const unsigned char* adj   = (const unsigned char*)d_in[2];
  const unsigned char* nmask = (const unsigned char*)d_in[3];
  /* d_in[4] node_roles: unused ('local' router) */
  const int* ord_cur  = (const int*)d_in[5];
  const int* ord_dst  = (const int*)d_in[6];
  const unsigned char* smask = (const unsigned char*)d_in[7];
  const float* em_norm_g = (const float*)d_in[8];
  const float* em_norm_b = (const float*)d_in[9];
  const float* em_src_w  = (const float*)d_in[10];
  const float* em_src_b  = (const float*)d_in[11];
  const float* em_g1_w   = (const float*)d_in[12];
  const float* em_g1_b   = (const float*)d_in[13];
  const float* em_g2_w   = (const float*)d_in[14];
  const float* em_g2_b   = (const float*)d_in[15];
  const float* em_out_w  = (const float*)d_in[16];
  const float* em_out_b  = (const float*)d_in[17];
  const float* sc_norm_g = (const float*)d_in[18];
  const float* sc_norm_b = (const float*)d_in[19];
  const float* sc_delta_w= (const float*)d_in[20];
  const float* sc_delta_b= (const float*)d_in[21];
  const float* sc_b_w    = (const float*)d_in[22];
  const float* sc_b_b    = (const float*)d_in[23];
  const float* sc_c_w    = (const float*)d_in[24];
  const float* sc_c_b    = (const float*)d_in[25];
  const float* sc_gate_w = (const float*)d_in[26];
  const float* sc_gate_b = (const float*)d_in[27];
  const float* sc_out_w  = (const float*)d_in[28];
  const float* sc_out_b  = (const float*)d_in[29];
  const float* sc_skip   = (const float*)d_in[30];
  const float* sc_log_a  = (const float*)d_in[31];
  const float* mg_w      = (const float*)d_in[32];
  const float* mg_b      = (const float*)d_in[33];
  const float* ff_norm_g = (const float*)d_in[34];
  const float* ff_norm_b = (const float*)d_in[35];
  const float* ff1_w     = (const float*)d_in[36];
  const float* ff1_b     = (const float*)d_in[37];
  const float* ff2_w     = (const float*)d_in[38];
  const float* ff2_b     = (const float*)d_in[39];

  // workspace layout (floats); total ~28.8 MB
  float* ws     = (float*)d_ws;
  float* normed = ws +  (size_t)0 * BND;
  float* srcb   = ws +  (size_t)1 * BND;
  float* prei   = ws +  (size_t)2 * BND;
  float* prej   = ws +  (size_t)3 * BND;
  float* msgs   = ws +  (size_t)4 * BND;
  float* x1     = ws +  (size_t)5 * BND;
  float* t1     = ws +  (size_t)6 * BND;
  float* xs     = ws +  (size_t)7 * BND;
  float* nrm    = ws +  (size_t)8 * BND;
  float* delta  = ws +  (size_t)9 * BND;
  float* gateb  = ws + (size_t)10 * BND;
  float* ygb    = ws + (size_t)11 * BND;
  float* xo     = ws + (size_t)12 * BND;
  float* curo   = ws + (size_t)13 * BND;
  float* dsto   = ws + (size_t)14 * BND;
  float* m1b    = ws + (size_t)15 * BND;
  float* mergeb = ws + (size_t)16 * BND;
  float* x2     = ws + (size_t)17 * BND;
  float* n2     = ws + (size_t)18 * BND;
  float* btb    = ws + (size_t)19 * BND;         // [B,N,D,DS]
  float* ctb    = btb + (size_t)BND * DSt;
  float* h1     = ctb + (size_t)BND * DSt;       // [B,N,4D]

  (void)hipFuncSetAttribute(reinterpret_cast<const void*>(&k_edge),
                            hipFuncAttributeMaxDynamicSharedMemorySize, EDGE_SMEM);

  dim3 blk(256);
  auto gemm = [&](const float* A, int lda, const float* W, int ldw, const float* bias,
                  const float* add, float* C, int M, int N, int K, int act) {
    k_gemm<<<dim3(N / 64, M / 64), blk, 0, stream>>>(A, lda, W, ldw, bias, add, C, M, N, K, act);
  };

  // ---- edge mixer ----
  k_ln<<<128, blk, 0, stream>>>(x, em_norm_g, em_norm_b, normed, ROWS);
  gemm(normed, Dm, em_src_w, Dm, em_src_b, nullptr, srcb, ROWS, Dm, Dm, 0);
  gemm(normed, Dm, em_g1_w,            Dm, nullptr, nullptr, prei, ROWS, Dm, Dm, 0);   // recv proj (W1 rows 0..127)
  gemm(srcb,   Dm, em_g1_w + 128 * Dm, Dm, em_g1_b, nullptr, prej, ROWS, Dm, Dm, 0);   // send proj + bias
  k_edge<<<64, blk, EDGE_SMEM, stream>>>(prei, prej, srcb, ef, adj, nmask,
                                         em_g1_w + 256 * Dm, em_g2_w, em_g2_b, msgs);
  gemm(x,    Dm, em_out_w,            Dm, em_out_b, x,  t1, ROWS, Dm, Dm, 0);          // x + x@Wtop + b
  gemm(msgs, Dm, em_out_w + 128 * Dm, Dm, nullptr,  t1, x1, ROWS, Dm, Dm, 0);          // + msg@Wbot -> x1

  // ---- two ordered Mamba-like scans ----
  for (int o = 0; o < 2; ++o) {
    const int* ord = o ? ord_dst : ord_cur;
    float* outb = o ? dsto : curo;
    k_gather<<<BND / 256, blk, 0, stream>>>(x1, ord, xs);
    k_ln<<<128, blk, 0, stream>>>(xs, sc_norm_g, sc_norm_b, nrm, ROWS);
    gemm(nrm, Dm, sc_delta_w, Dm,        sc_delta_b, nullptr, delta, ROWS, Dm,        Dm, 3);
    gemm(nrm, Dm, sc_b_w,     Dm * DSt,  sc_b_b,     nullptr, btb,   ROWS, Dm * DSt,  Dm, 0);
    gemm(nrm, Dm, sc_c_w,     Dm * DSt,  sc_c_b,     nullptr, ctb,   ROWS, Dm * DSt,  Dm, 0);
    gemm(nrm, Dm, sc_gate_w,  Dm,        sc_gate_b,  nullptr, gateb, ROWS, Dm,        Dm, 2);
    k_scan<<<8, dim3(32), 0, stream>>>(delta, btb, ctb, gateb, nrm, sc_skip, sc_log_a, smask, ygb);
    gemm(ygb, Dm, sc_out_w, Dm, sc_out_b, xs, xo, ROWS, Dm, Dm, 0);                    // xs + lin(gate*y)
    k_scatter<<<BND / 256, blk, 0, stream>>>(xo, ord, outb);
  }

  // ---- merge gate ----
  gemm(x1,   Dm, mg_w,            Dm, mg_b,    nullptr, m1b,    ROWS, Dm, Dm, 0);
  gemm(curo, Dm, mg_w + 128 * Dm, Dm, nullptr, m1b,     t1,     ROWS, Dm, Dm, 0);
  gemm(dsto, Dm, mg_w + 256 * Dm, Dm, nullptr, t1,      mergeb, ROWS, Dm, Dm, 2);
  k_combine<<<BND / 256, blk, 0, stream>>>(x1, mergeb, curo, dsto, x2);

  // ---- FFN ----
  k_ln<<<128, blk, 0, stream>>>(x2, ff_norm_g, ff_norm_b, n2, ROWS);
  gemm(n2, Dm,     ff1_w, 4 * Dm, ff1_b, nullptr, h1,             ROWS, 4 * Dm, Dm,     1);
  gemm(h1, 4 * Dm, ff2_w, Dm,     ff2_b, x2,      (float*)d_out,  ROWS, Dm,     4 * Dm, 0);
}
